// PhysicsGNN_46351287058883
// MI455X (gfx1250) — matrix operand order repaired and checked by reading, not compiled, not run on hardware
//
#include <hip/hip_runtime.h>

#define NBATCH 4
#define NN 512
#define HH 64
#define LEPS 1e-5f

typedef __attribute__((ext_vector_type(16))) __bf16 v16bf;
typedef __attribute__((ext_vector_type(8)))  float  v8f;
typedef unsigned short us8 __attribute__((ext_vector_type(8)));

union ABfrag { v16bf v; unsigned short s[16]; us8 h[2]; };
union F8     { v8f v; float f[8]; };

__device__ __forceinline__ unsigned short f2bf(float x) {
    unsigned int u = __float_as_uint(x);
    u += 0x7FFFu + ((u >> 16) & 1u);          // round-to-nearest-even
    return (unsigned short)(u >> 16);
}

// ---------------- K1: nf = LN(relu(x @ enc_W + enc_b)) ----------------
__global__ __launch_bounds__(64) void k_encoder(
    const float* __restrict__ x, const float* __restrict__ W,
    const float* __restrict__ b, const float* __restrict__ g,
    const float* __restrict__ be, float* __restrict__ nf)
{
    int row = blockIdx.x, f = threadIdx.x;
    __shared__ float sx[16];
    __shared__ float s1[64], s2[64];
    if (f < 16) sx[f] = x[row * 16 + f];
    __syncthreads();
    float acc = b[f];
#pragma unroll
    for (int k = 0; k < 16; ++k) acc += sx[k] * W[k * 64 + f];
    float r = fmaxf(acc, 0.f);
    s1[f] = r; s2[f] = r * r;
    __syncthreads();
    for (int s = 32; s > 0; s >>= 1) {
        if (f < s) { s1[f] += s1[f + s]; s2[f] += s2[f + s]; }
        __syncthreads();
    }
    float mu = s1[0] * (1.f / 64.f);
    float var = s2[0] * (1.f / 64.f) - mu * mu;
    nf[row * 64 + f] = (r - mu) * rsqrtf(var + LEPS) * g[f] + be[f];
}

// ---------------- K2: mi = nf@Wi, mj = nf@Wj ----------------
__global__ __launch_bounds__(128) void k_mimj(
    const float* __restrict__ nf, const float* __restrict__ msg_W, int l,
    float* __restrict__ mi, float* __restrict__ mj)
{
    int row = blockIdx.x, t = threadIdx.x;
    __shared__ float snf[64];
    if (t < 64) snf[t] = nf[row * 64 + t];
    __syncthreads();
    int f = t & 63, sel = t >> 6;
    const float* Wp = msg_W + (size_t)(l * 192 + sel * 64) * 64;
    float acc = 0.f;
#pragma unroll 8
    for (int k = 0; k < 64; ++k) acc += snf[k] * Wp[k * 64 + f];
    (sel ? mj : mi)[row * 64 + f] = acc;
}

// ---------------- K3: fused edge pipeline (WMMA bf16) ----------------
// msg[b,i] = mean_j LN(relu(mi[i] + mj[j] + LN(relu((pos_i-pos_j)@eW+eb))@We + mb))
__global__ __launch_bounds__(128) void k_msg(
    const float* __restrict__ x,
    const float* __restrict__ mi, const float* __restrict__ mj,
    const float* __restrict__ edge_W, const float* __restrict__ edge_b,
    const float* __restrict__ edge_g, const float* __restrict__ edge_be,
    const float* __restrict__ msg_W, const float* __restrict__ msg_b,
    const float* __restrict__ msg_g, const float* __restrict__ msg_be,
    int l, float* __restrict__ msg_out)
{
    __shared__ unsigned short sWe[64 * 64];                    // We as bf16
    __shared__ __align__(16) unsigned short sA[4][16 * 72];    // A staging, padded rows
    __shared__ float sE[5][64];                                // edge W0,W1,b,g,be
    __shared__ float sAcc[4][64];

    const int bi = blockIdx.x;              // b*512 + i
    const int b  = bi >> 9;
    const int tid = threadIdx.x;
    const int wv = tid >> 5, ln = tid & 31;
    const int hf = ln >> 4, lq = ln & 15;
    const int rowbase = b * NN;

    // stage We (rows 128..191 of msg_W[l]) as bf16
    for (int idx = tid; idx < 64 * 64; idx += 128) {
        int h = idx >> 6, k = idx & 63;
        sWe[idx] = f2bf(msg_W[(size_t)(l * 192 + 128 + h) * 64 + k]);
    }
    if (tid < 64) {
        sE[0][tid] = edge_W[tid];
        sE[1][tid] = edge_W[64 + tid];
        sE[2][tid] = edge_b[tid];
        sE[3][tid] = edge_g[tid];
        sE[4][tid] = edge_be[tid];
    }
    __syncthreads();

    // per-lane constants: feature f = nb*16 + lq
    float mib[4], gv[4], bev[4];
#pragma unroll
    for (int nb = 0; nb < 4; ++nb) {
        int f = nb * 16 + lq;
        mib[nb] = mi[(size_t)bi * 64 + f] + msg_b[l * 64 + f];
        gv[nb]  = msg_g[l * 64 + f];
        bev[nb] = msg_be[l * 64 + f];
    }
    const float pix = x[(size_t)bi * 16 + 14];
    const float piy = x[(size_t)bi * 16 + 15];

    // B fragments: 32x16 bf16 layout -> lane<16: K0-7 & 16-23 ; lane>=16: K8-15 & 24-31
    ABfrag bfrag[2][4];
#pragma unroll
    for (int kb = 0; kb < 2; ++kb)
#pragma unroll
        for (int nb = 0; nb < 4; ++nb) {
#pragma unroll
            for (int e = 0; e < 16; ++e) {
                int kk = (e < 8) ? (hf * 8 + e) : (16 + hf * 8 + (e - 8));
                bfrag[kb][nb].s[e] = sWe[(kb * 32 + kk) * 64 + nb * 16 + lq];
            }
        }

    float acc[4] = {0.f, 0.f, 0.f, 0.f};

    for (int jt = wv; jt < 32; jt += 4) {
        const int j0 = jt * 16;
        // ---- edge features for row (j = j0+lq), feature half hf ----
        const int jr = j0 + lq;
        float dx = pix - x[(size_t)(rowbase + jr) * 16 + 14];
        float dy = piy - x[(size_t)(rowbase + jr) * 16 + 15];
        float rv[32], sum = 0.f, ssq = 0.f;
#pragma unroll
        for (int t = 0; t < 32; ++t) {
            int f = hf * 32 + t;
            float v = fmaxf(dx * sE[0][f] + dy * sE[1][f] + sE[2][f], 0.f);
            rv[t] = v; sum += v; ssq += v * v;
        }
        sum += __shfl_xor(sum, 16, 32);
        ssq += __shfl_xor(ssq, 16, 32);
        float mu = sum * (1.f / 64.f);
        float inv = rsqrtf(ssq * (1.f / 64.f) - mu * mu + LEPS);
        union { unsigned short s[32]; us8 h[4]; } ot;
#pragma unroll
        for (int t = 0; t < 32; ++t) {
            int f = hf * 32 + t;
            ot.s[t] = f2bf((rv[t] - mu) * inv * sE[3][f] + sE[4][f]);
        }
#pragma unroll
        for (int c = 0; c < 4; ++c)
            *reinterpret_cast<us8*>(&sA[wv][lq * 72 + hf * 32 + c * 8]) = ot.h[c];
        asm volatile("s_wait_dscnt 0" ::: "memory");

        // ---- WMMA: (16 x 64) ef-tile @ (64 x 64) We ----
        F8 d[4];
#pragma unroll
        for (int nb = 0; nb < 4; ++nb)
#pragma unroll
            for (int q = 0; q < 8; ++q) d[nb].f[q] = 0.f;
#pragma unroll
        for (int kb = 0; kb < 2; ++kb) {
            // A 16x32 bf16 layout: lane<16 -> K0-7 & 16-23 ; lane>=16 -> K8-15 & 24-31
            ABfrag a;
            const unsigned short* ap = &sA[wv][lq * 72 + kb * 32 + hf * 8];
            a.h[0] = *reinterpret_cast<const us8*>(ap);
            a.h[1] = *reinterpret_cast<const us8*>(ap + 16);
#pragma unroll
            for (int nb = 0; nb < 4; ++nb)
                d[nb].v = __builtin_amdgcn_wmma_f32_16x16x32_bf16(
                    false, a.v, false, bfrag[kb][nb].v, (short)0, d[nb].v,
                    false, false);
        }

        // ---- epilogue: +mi+mj+b, relu, per-row LN, accumulate mean_j ----
#pragma unroll
        for (int v = 0; v < 8; ++v) {
            const int jrow = j0 + hf * 8 + v;
            float s = 0.f, ss = 0.f;
#pragma unroll
            for (int nb = 0; nb < 4; ++nb) {
                float val = d[nb].f[v] + mib[nb] +
                            mj[(size_t)(rowbase + jrow) * 64 + nb * 16 + lq];
                val = fmaxf(val, 0.f);
                d[nb].f[v] = val;
                s += val; ss += val * val;
            }
#pragma unroll
            for (int m = 1; m < 16; m <<= 1) {
                s  += __shfl_xor(s,  m, 32);
                ss += __shfl_xor(ss, m, 32);
            }
            float mu2 = s * (1.f / 64.f);
            float inv2 = rsqrtf(ss * (1.f / 64.f) - mu2 * mu2 + LEPS);
#pragma unroll
            for (int nb = 0; nb < 4; ++nb)
                acc[nb] += (d[nb].f[v] - mu2) * inv2 * gv[nb] + bev[nb];
        }
    }

    // combine halves, then waves
#pragma unroll
    for (int nb = 0; nb < 4; ++nb) {
        float tot = acc[nb] + __shfl_xor(acc[nb], 16, 32);
        if (hf == 0) sAcc[wv][nb * 16 + lq] = tot;
    }
    __syncthreads();
    if (tid < 64) {
        float m = (sAcc[0][tid] + sAcc[1][tid] + sAcc[2][tid] + sAcc[3][tid]) *
                  (1.f / 512.f);
        msg_out[(size_t)bi * 64 + tid] = m;
    }
}

// ---------------- K4: nf' = LN(relu([nf|msg] @ upd_W + upd_b)) ----------------
__global__ __launch_bounds__(64) void k_update(
    const float* __restrict__ nf, const float* __restrict__ msg,
    const float* __restrict__ W, const float* __restrict__ b,
    const float* __restrict__ g, const float* __restrict__ be,
    int l, float* __restrict__ nf_out)
{
    int row = blockIdx.x, f = threadIdx.x;
    __shared__ float cat[128];
    __shared__ float s1[64], s2[64];
    cat[f]      = nf[row * 64 + f];
    cat[64 + f] = msg[row * 64 + f];
    __syncthreads();
    float acc = b[l * 64 + f];
    const float* Wp = W + (size_t)l * 128 * 64;
#pragma unroll 8
    for (int k = 0; k < 128; ++k) acc += cat[k] * Wp[k * 64 + f];
    float r = fmaxf(acc, 0.f);
    s1[f] = r; s2[f] = r * r;
    __syncthreads();
    for (int s = 32; s > 0; s >>= 1) {
        if (f < s) { s1[f] += s1[f + s]; s2[f] += s2[f + s]; }
        __syncthreads();
    }
    float mu = s1[0] * (1.f / 64.f);
    float var = s2[0] * (1.f / 64.f) - mu * mu;
    nf_out[row * 64 + f] = (r - mu) * rsqrtf(var + LEPS) * g[l * 64 + f] + be[l * 64 + f];
}

// ---------------- K5: out = x + nf @ out_W + out_b ----------------
__global__ __launch_bounds__(64) void k_output(
    const float* __restrict__ x, const float* __restrict__ nf,
    const float* __restrict__ W, const float* __restrict__ b,
    float* __restrict__ out)
{
    int t = threadIdx.x;
    __shared__ float snf[256];
    for (int idx = t; idx < 256; idx += 64)
        snf[idx] = nf[(size_t)blockIdx.x * 256 + idx];
    __syncthreads();
    int r4 = t >> 4, c = t & 15;
    int row = blockIdx.x * 4 + r4;
    float acc = x[row * 16 + c] + b[c];
#pragma unroll 8
    for (int k = 0; k < 64; ++k) acc += snf[r4 * 64 + k] * W[k * 16 + c];
    out[row * 16 + c] = acc;
}

extern "C" void kernel_launch(void* const* d_in, const int* in_sizes, int n_in,
                              void* d_out, int out_size, void* d_ws, size_t ws_size,
                              hipStream_t stream) {
    const float* x      = (const float*)d_in[0];
    const float* enc_W  = (const float*)d_in[1];
    const float* enc_b  = (const float*)d_in[2];
    const float* enc_g  = (const float*)d_in[3];
    const float* enc_be = (const float*)d_in[4];
    const float* edge_W = (const float*)d_in[5];
    const float* edge_b = (const float*)d_in[6];
    const float* edge_g = (const float*)d_in[7];
    const float* edge_be= (const float*)d_in[8];
    const float* msg_W  = (const float*)d_in[9];
    const float* msg_b  = (const float*)d_in[10];
    const float* msg_g  = (const float*)d_in[11];
    const float* msg_be = (const float*)d_in[12];
    const float* upd_W  = (const float*)d_in[13];
    const float* upd_b  = (const float*)d_in[14];
    const float* upd_g  = (const float*)d_in[15];
    const float* upd_be = (const float*)d_in[16];
    const float* out_W  = (const float*)d_in[17];
    const float* out_b  = (const float*)d_in[18];
    float* out = (float*)d_out;

    const int ROWS = NBATCH * NN;             // 2048
    float* nf_a = (float*)d_ws;
    float* nf_b = nf_a + (size_t)ROWS * 64;
    float* mi   = nf_b + (size_t)ROWS * 64;
    float* mj   = mi   + (size_t)ROWS * 64;
    float* msg  = mj   + (size_t)ROWS * 64;

    k_encoder<<<ROWS, 64, 0, stream>>>(x, enc_W, enc_b, enc_g, enc_be, nf_a);

    for (int l = 0; l < 3; ++l) {
        const float* nin = (l & 1) ? nf_b : nf_a;
        float* nout      = (l & 1) ? nf_a : nf_b;
        k_mimj<<<ROWS, 128, 0, stream>>>(nin, msg_W, l, mi, mj);
        k_msg<<<ROWS, 128, 0, stream>>>(x, mi, mj, edge_W, edge_b, edge_g, edge_be,
                                        msg_W, msg_b, msg_g, msg_be, l, msg);
        k_update<<<ROWS, 64, 0, stream>>>(nin, msg, upd_W, upd_b, upd_g, upd_be, l, nout);
    }
    k_output<<<ROWS / 4, 64, 0, stream>>>(x, nf_b, out_W, out_b, out);
}
